// BiologicalMemory_55499567398938
// MI455X (gfx1250) — matrix operand order repaired and checked by reading, not compiled, not run on hardware
//
#include <hip/hip_runtime.h>
#include <stdint.h>

// ---------------------------------------------------------------------------
// CDNA5 (gfx1250) wave32 WMMA types
// ---------------------------------------------------------------------------
typedef __attribute__((ext_vector_type(16))) __bf16        v16bf;
typedef __attribute__((ext_vector_type(8)))  float         v8f;
typedef __attribute__((ext_vector_type(4)))  unsigned int  v4u;
typedef __attribute__((ext_vector_type(4)))  float         v4f;

union BF16Frag {
    v16bf bf;     // WMMA operand (16 bf16 per lane)
    v4u   q[2];   // two 16-byte chunks for b128 loads
};

#define EPS 1e-8f
#define SIM_THRESHOLD 0.6f

__device__ __forceinline__ unsigned short f2bf(float f) {
    unsigned int u = __float_as_uint(f);
    unsigned int r = u + 0x7FFFu + ((u >> 16) & 1u);   // round-to-nearest-even
    return (unsigned short)(r >> 16);
}

// order-preserving float -> u32 key (larger float => larger key)
__device__ __forceinline__ unsigned int fkey(float f) {
    unsigned int u = __float_as_uint(f);
    return (u & 0x80000000u) ? ~u : (u | 0x80000000u);
}

// async global -> LDS copy of one 16B chunk (CDNA5, tracked by ASYNCcnt)
__device__ __forceinline__ void async_g2l_b128(unsigned int lds_addr,
                                               const void* gptr) {
    asm volatile("global_load_async_to_lds_b128 %0, %1, off"
                 :: "v"(lds_addr), "v"(gptr)
                 : "memory");
}
__device__ __forceinline__ void wait_asynccnt0() {
    asm volatile("s_wait_asynccnt 0x0" ::: "memory");
}

// ---------------------------------------------------------------------------
// Kernel 0: zero the global best[] keys (must be re-init every call)
// ---------------------------------------------------------------------------
__global__ void init_best_kernel(unsigned long long* g) {
    g[threadIdx.x] = 0ULL;
}

// ---------------------------------------------------------------------------
// Kernel 1: fp32 [rows x 1024] -> bf16 copy + per-row L2 norm
// one block (256 thr) per row, 4 elements/thread
// ---------------------------------------------------------------------------
__global__ void __launch_bounds__(256)
prep_kernel(const float* __restrict__ src, unsigned short* __restrict__ dst,
            float* __restrict__ norms, int cols)
{
    const int row = blockIdx.x;
    const int tid = threadIdx.x;
    const float* s = src + (size_t)row * cols;
    unsigned short* d = dst + (size_t)row * cols;

    v4f v = *(const v4f*)(s + tid * 4);
    float ss = v.x * v.x + v.y * v.y + v.z * v.z + v.w * v.w;

    unsigned long long pk =
        (unsigned long long)f2bf(v.x)
      | ((unsigned long long)f2bf(v.y) << 16)
      | ((unsigned long long)f2bf(v.z) << 32)
      | ((unsigned long long)f2bf(v.w) << 48);
    *(unsigned long long*)(d + tid * 4) = pk;

    #pragma unroll
    for (int o = 16; o > 0; o >>= 1) ss += __shfl_xor(ss, o, 32);

    __shared__ float wsum[8];
    if ((tid & 31) == 0) wsum[tid >> 5] = ss;
    __syncthreads();
    if (tid == 0) {
        float t = 0.f;
        #pragma unroll
        for (int i = 0; i < 8; ++i) t += wsum[i];
        norms[row] = sqrtf(t);
    }
}

// ---------------------------------------------------------------------------
// Kernel 2: bf16 WMMA GEMM (memories x query^T) + cosine scale + argmax.
// WG = 256 thr = 8 waves. WG tile: 128 memory rows x 256 queries.
// Query K-slices double-buffered in LDS via async global->LDS copies
// (ASYNCcnt); A fragments and B fragments software-pipelined.
// ---------------------------------------------------------------------------
__global__ void __launch_bounds__(256)
sim_argmax_kernel(const unsigned short* __restrict__ memb,
                  const unsigned short* __restrict__ qb,
                  const float* __restrict__ mnorm,
                  const float* __restrict__ qnorm,
                  unsigned long long* __restrict__ gbest,
                  int N, int D)
{
    __shared__ unsigned short qs[2][256][40];  // 32 used, pad->40 (80B rows)
    __shared__ unsigned long long sbest[256];
    __shared__ float qn[256];

    const int tid  = threadIdx.x;
    const int lane = tid & 31;
    const int wave = tid >> 5;       // 0..7
    const int lo   = lane & 15;
    const int hi   = lane >> 4;      // 0 or 1

    sbest[tid] = 0ULL;
    qn[tid]    = qnorm[tid];

    const int tileBase = blockIdx.x * 128;
    const int rowBase  = tileBase + wave * 16;

    // A fragment source: this lane's memory row (clamped for the tail tile)
    const int arow  = rowBase + lo;
    const int arowc = arow < N ? arow : (N - 1);
    const unsigned short* aptr = memb + (size_t)arowc * D + hi * 16;

    // norms of the 8 C-matrix rows this lane owns (M = r + 8*hi)
    const int mbase = rowBase + 8 * hi;
    float mn[8];
    #pragma unroll
    for (int r = 0; r < 8; ++r) {
        int mr = mbase + r;
        mn[r] = mnorm[mr < N ? mr : (N - 1)];
    }

    // async stage: 256 threads each copy their query row's 32-bf16 K-slice
    const unsigned short* qsrc = qb + (size_t)tid * D;
    const unsigned int lds_row[2] = {
        (unsigned int)(unsigned long long)&qs[0][tid][0],
        (unsigned int)(unsigned long long)&qs[1][tid][0]
    };
    auto stage = [&](int buf, int kk) {
        const unsigned short* g = qsrc + kk;
        #pragma unroll
        for (int c = 0; c < 4; ++c)
            async_g2l_b128(lds_row[buf] + c * 16, g + c * 8);
    };

    v8f acc[16] = {};

    // prologue: stage slice 0, preload A fragment for kk=0
    stage(0, 0);
    BF16Frag a_cur;
    a_cur.q[0] = *(const v4u*)(aptr);
    a_cur.q[1] = *(const v4u*)(aptr + 8);
    wait_asynccnt0();
    __syncthreads();

    for (int kk = 0; kk < D; kk += 32) {
        const int buf = (kk >> 5) & 1;

        // kick off async stage of the next slice into the other buffer
        if (kk + 32 < D) stage(buf ^ 1, kk + 32);

        // preload next A fragment (global) so loads overlap the WMMA burst
        BF16Frag a_nxt = a_cur;
        if (kk + 32 < D) {
            a_nxt.q[0] = *(const v4u*)(aptr + kk + 32);
            a_nxt.q[1] = *(const v4u*)(aptr + kk + 32 + 8);
        }

        // software-pipelined B fragments from LDS
        BF16Frag bcur, bnxt;
        {
            const unsigned short* qrow0 = &qs[buf][lo][hi * 16];
            bcur.q[0] = *(const v4u*)(qrow0);
            bcur.q[1] = *(const v4u*)(qrow0 + 8);
        }
        bnxt = bcur;
        #pragma unroll
        for (int nt = 0; nt < 16; ++nt) {
            if (nt < 15) {
                const unsigned short* qrow = &qs[buf][(nt + 1) * 16 + lo][hi * 16];
                bnxt.q[0] = *(const v4u*)(qrow);
                bnxt.q[1] = *(const v4u*)(qrow + 8);
            }
            acc[nt] = __builtin_amdgcn_wmma_f32_16x16x32_bf16(
                false, a_cur.bf, false, bcur.bf, (short)0, acc[nt], false, false);
            bcur = bnxt;
        }
        a_cur = a_nxt;

        // next slice fully in LDS + all waves done reading this one
        wait_asynccnt0();
        __syncthreads();
    }

    // cosine scale + per-lane argmax over 8 rows, merge lane pair, LDS max
    #pragma unroll
    for (int nt = 0; nt < 16; ++nt) {
        const int q   = nt * 16 + lo;
        const float rq = qn[q];
        float bsim = -1e30f;
        int   bidx = 0;
        #pragma unroll
        for (int r = 0; r < 8; ++r) {
            int mr = mbase + r;
            float den = rq * mn[r];
            den = den > EPS ? den : EPS;
            float s = acc[nt][r] / den;
            bool ok = (mr < N) && (s > bsim);
            bsim = ok ? s : bsim;
            bidx = ok ? mr : bidx;
        }
        unsigned long long key =
            ((unsigned long long)fkey(bsim) << 32) | (unsigned int)bidx;
        unsigned long long oth = __shfl_xor(key, 16, 32);
        key = key > oth ? key : oth;
        if (hi == 0) atomicMax(&sbest[q], key);
    }
    __syncthreads();
    atomicMax(&gbest[tid], sbest[tid]);   // one global atomic per query per WG
}

// ---------------------------------------------------------------------------
// Kernel 3: unpack winner, threshold mask, gather winning bf16 row
// ---------------------------------------------------------------------------
__global__ void __launch_bounds__(256)
gather_kernel(const unsigned long long* __restrict__ gbest,
              const unsigned short* __restrict__ memb,
              unsigned short* __restrict__ emb,
              float* __restrict__ mask, int D)
{
    __shared__ int sidx;
    const int b = blockIdx.x;
    if (threadIdx.x == 0) {
        unsigned long long key = gbest[b];
        unsigned int uk = (unsigned int)(key >> 32);
        unsigned int fb = (uk & 0x80000000u) ? (uk & 0x7FFFFFFFu) : ~uk;
        float sim = __uint_as_float(fb);
        sidx = (int)(key & 0xFFFFFFFFu);
        mask[b] = (sim > SIM_THRESHOLD) ? 1.0f : 0.0f;
    }
    __syncthreads();
    const unsigned short* srcr = memb + (size_t)sidx * D;
    unsigned short* dstr = emb + (size_t)b * D;
    *(unsigned long long*)(dstr + threadIdx.x * 4) =
        *(const unsigned long long*)(srcr + threadIdx.x * 4);
}

// ---------------------------------------------------------------------------
// Kernel 4: decode GEMM out[256,1024] = E[256,1024] @ W^T + b, then mask.
// grid (16, 8): blockIdx.x = 16-row query tile; each of 8 waves owns one
// 16-col output tile. A staged in LDS (shared by all waves), B from global.
// ---------------------------------------------------------------------------
__global__ void __launch_bounds__(256)
decode_kernel(const unsigned short* __restrict__ emb,
              const unsigned short* __restrict__ wb,
              const float* __restrict__ bias,
              const float* __restrict__ mask,
              float* __restrict__ out, int D)
{
    __shared__ unsigned short es[16][40];
    const int tid  = threadIdx.x;
    const int lane = tid & 31;
    const int wave = tid >> 5;
    const int lo   = lane & 15;
    const int hi   = lane >> 4;

    const int mbase = blockIdx.x * 16;
    const int nbase = (blockIdx.y * 8 + wave) * 16;

    const unsigned short* wrow = wb + (size_t)(nbase + lo) * D;
    v8f acc = {};

    for (int kk = 0; kk < D; kk += 32) {
        __syncthreads();
        if (tid < 64) {                      // stage 16x32 bf16 A-slice
            int r = tid >> 2, c = tid & 3;
            *(v4u*)&es[r][c * 8] =
                *(const v4u*)(emb + (size_t)(mbase + r) * D + kk + c * 8);
        }
        __syncthreads();

        BF16Frag a, b;
        a.q[0] = *(const v4u*)&es[lo][hi * 16];
        a.q[1] = *(const v4u*)&es[lo][hi * 16 + 8];
        b.q[0] = *(const v4u*)(wrow + kk + hi * 16);
        b.q[1] = *(const v4u*)(wrow + kk + hi * 16 + 8);
        acc = __builtin_amdgcn_wmma_f32_16x16x32_bf16(
            false, a.bf, false, b.bf, (short)0, acc, false, false);
    }

    const int n = nbase + lo;
    const float bz = bias[n];
    #pragma unroll
    for (int r = 0; r < 8; ++r) {
        int m = mbase + r + 8 * hi;
        out[(size_t)m * D + n] = (acc[r] + bz) * mask[m];
    }
}

// ---------------------------------------------------------------------------
extern "C" void kernel_launch(void* const* d_in, const int* in_sizes, int n_in,
                              void* d_out, int out_size, void* d_ws, size_t ws_size,
                              hipStream_t stream)
{
    const float* query    = (const float*)d_in[0];
    const float* memories = (const float*)d_in[1];
    const float* dec_w    = (const float*)d_in[2];
    const float* dec_b    = (const float*)d_in[3];

    const int B = 256, D = 1024;
    const int N = in_sizes[1] / D;   // 50000

    char* ws = (char*)d_ws;
    size_t off = 0;
    auto wsalloc = [&](size_t bytes) -> void* {
        void* p = ws + off;
        off += (bytes + 255) & ~(size_t)255;
        return p;
    };

    unsigned short*     memb  = (unsigned short*)wsalloc((size_t)N * D * 2);
    unsigned short*     qb    = (unsigned short*)wsalloc((size_t)B * D * 2);
    unsigned short*     wbf   = (unsigned short*)wsalloc((size_t)D * D * 2);
    float*              mnorm = (float*)wsalloc((size_t)N * 4);
    float*              qnorm = (float*)wsalloc((size_t)B * 4);
    float*              wnorm = (float*)wsalloc((size_t)D * 4);   // scratch
    unsigned long long* gbest = (unsigned long long*)wsalloc((size_t)B * 8);
    unsigned short*     emb   = (unsigned short*)wsalloc((size_t)B * D * 2);
    float*              mask  = (float*)wsalloc((size_t)B * 4);

    init_best_kernel<<<1, B, 0, stream>>>(gbest);
    prep_kernel<<<N, 256, 0, stream>>>(memories, memb, mnorm, D);
    prep_kernel<<<B, 256, 0, stream>>>(query, qb, qnorm, D);
    prep_kernel<<<D, 256, 0, stream>>>(dec_w, wbf, wnorm, D);

    sim_argmax_kernel<<<(N + 127) / 128, 256, 0, stream>>>(
        memb, qb, mnorm, qnorm, gbest, N, D);

    gather_kernel<<<B, 256, 0, stream>>>(gbest, memb, emb, mask, D);

    decode_kernel<<<dim3(B / 16, D / 128), 256, 0, stream>>>(
        emb, wbf, dec_b, mask, (float*)d_out, D);
}
